// MultiHeadDotProductAttention_52158082843254
// MI455X (gfx1250) — compile-verified, matrix operand order checked by reading, not run on hardware
//
#include <hip/hip_runtime.h>

typedef unsigned int   u32;
typedef unsigned short u16;

typedef __attribute__((ext_vector_type(16))) __bf16 bf16x16;
typedef __attribute__((ext_vector_type(8)))  float  f32x8;
typedef __attribute__((ext_vector_type(4)))  float  f32x4;
typedef __attribute__((ext_vector_type(4)))  u32    u32x4;

#define NB   4
#define NT   2048
#define NH   16
#define ND   64
#define NE   1024
#define NM   (NB*NT)      // 8192 rows
#define NSB  (NT/32)      // 64 S-blocks

union Frag {
  bf16x16 v;
  u32x4   q[2];
};

__device__ __forceinline__ f32x8 zero8() {
  f32x8 z = {0.f,0.f,0.f,0.f,0.f,0.f,0.f,0.f};
  return z;
}

__device__ __forceinline__ u16 f2bf(float f) {
  u32 u = __float_as_uint(f);
  u32 r = (u + 0x7FFFu + ((u >> 16) & 1u)) >> 16;   // RNE
  return (u16)r;
}

__device__ __forceinline__ u32x4 pack8(const float* __restrict__ p) {
  f32x4 a = *(const f32x4*)p;
  f32x4 b = *(const f32x4*)(p + 4);
  u32x4 r;
  r.x = (u32)f2bf(a.x) | ((u32)f2bf(a.y) << 16);
  r.y = (u32)f2bf(a.z) | ((u32)f2bf(a.w) << 16);
  r.z = (u32)f2bf(b.x) | ((u32)f2bf(b.y) << 16);
  r.w = (u32)f2bf(b.z) | ((u32)f2bf(b.w) << 16);
  return r;
}

__device__ __forceinline__ f32x8 wmma_bf16(const Frag& a, const Frag& b, f32x8 c) {
  return __builtin_amdgcn_wmma_f32_16x16x32_bf16(
      false, a.v, false, b.v, (short)0, c, false, false);
}

// Async DMA: global -> LDS, 16B per lane, tracked by ASYNCcnt.
__device__ __forceinline__ void async_b128(unsigned lds_off, unsigned long long gaddr) {
  asm volatile("global_load_async_to_lds_b128 %0, %1, off"
               :: "v"(lds_off), "v"(gaddr) : "memory");
}

// ---------------- fp32 -> bf16 conversion (weights) ----------------
__global__ void cvt_bf16_kernel(const float* __restrict__ in, u16* __restrict__ out, int n) {
  int i = blockIdx.x * 256 + threadIdx.x;
  if (i < n) out[i] = f2bf(in[i]);
}

// ---------------- projection + layernorm (q, k) ----------------
// block = 256 (8 waves); block computes 16 rows x 1024 cols, wave handles 16x128.
__global__ __launch_bounds__(256) void proj_ln_kernel(
    const float* __restrict__ X,     // [NM, NE] fp32
    const u16*   __restrict__ Wbf,   // [NE, NE] bf16 (row = out-feature)
    const float* __restrict__ gamma,
    const float* __restrict__ beta,
    u16*         __restrict__ Obf)   // [NM, NE] bf16
{
  __shared__ float tile[16][NE];     // 64 KB
  const int m0   = blockIdx.x * 16;
  const int tid  = threadIdx.x;
  const int wave = tid >> 5;
  const int lane = tid & 31;
  const int ln   = lane & 15;
  const int half = lane >> 4;
  const int n_base = wave * 128;

  f32x8 acc[8];
#pragma unroll
  for (int i = 0; i < 8; i++) acc[i] = zero8();

  const float* xrow = X + (size_t)(m0 + ln) * NE;

  for (int k0 = 0; k0 < NE; k0 += 32) {
    const int c0 = k0 + half * 8;
    __builtin_prefetch(xrow + c0 + 64);            // next A panel -> global_prefetch_b8
    Frag a;
    a.q[0] = pack8(xrow + c0);
    a.q[1] = pack8(xrow + c0 + 16);
#pragma unroll
    for (int nt = 0; nt < 8; nt++) {
      const u16* wrow = Wbf + (size_t)(n_base + nt * 16 + ln) * NE + c0;
      Frag b;
      b.q[0] = *(const u32x4*)(wrow);
      b.q[1] = *(const u32x4*)(wrow + 16);
      acc[nt] = wmma_bf16(a, b, acc[nt]);
    }
  }

  // spill tiles to LDS (C-layout: col = ln, rows r + 8*half)
#pragma unroll
  for (int nt = 0; nt < 8; nt++)
#pragma unroll
    for (int r = 0; r < 8; r++)
      tile[r + 8 * half][n_base + nt * 16 + ln] = acc[nt][r];
  __syncthreads();

  // layernorm: 16 threads per row
  const int row = tid >> 4;
  const int cth = tid & 15;
  float s1 = 0.f, s2 = 0.f;
  for (int j = 0; j < 64; j++) {
    float v = tile[row][cth + j * 16];
    s1 += v; s2 += v * v;
  }
#pragma unroll
  for (int m = 1; m < 16; m <<= 1) {
    s1 += __shfl_xor(s1, m, 32);
    s2 += __shfl_xor(s2, m, 32);
  }
  float mu  = s1 * (1.f / NE);
  float var = s2 * (1.f / NE) - mu * mu;
  float rs  = rsqrtf(var + 1e-5f);
  u16* orow = Obf + (size_t)(m0 + row) * NE;
  for (int j = 0; j < 64; j++) {
    int col = cth + j * 16;
    float v = (tile[row][col] - mu) * rs * gamma[col] + beta[col];
    orow[col] = f2bf(v);
  }
}

// ---------------- projection (v), writes transposed per head ----------------
__global__ __launch_bounds__(256) void proj_v_kernel(
    const float* __restrict__ X,     // [NM, NE]
    const u16*   __restrict__ Wbf,   // [NE, NE]
    u16*         __restrict__ vT)    // [NB*NH*ND, NT]
{
  const int m0   = blockIdx.x * 16;
  const int tid  = threadIdx.x;
  const int wave = tid >> 5;
  const int lane = tid & 31;
  const int ln   = lane & 15;
  const int half = lane >> 4;
  const int n_base = wave * 128;

  f32x8 acc[8];
#pragma unroll
  for (int i = 0; i < 8; i++) acc[i] = zero8();

  const float* xrow = X + (size_t)(m0 + ln) * NE;

  for (int k0 = 0; k0 < NE; k0 += 32) {
    const int c0 = k0 + half * 8;
    __builtin_prefetch(xrow + c0 + 64);
    Frag a;
    a.q[0] = pack8(xrow + c0);
    a.q[1] = pack8(xrow + c0 + 16);
#pragma unroll
    for (int nt = 0; nt < 8; nt++) {
      const u16* wrow = Wbf + (size_t)(n_base + nt * 16 + ln) * NE + c0;
      Frag b;
      b.q[0] = *(const u32x4*)(wrow);
      b.q[1] = *(const u32x4*)(wrow + 16);
      acc[nt] = wmma_bf16(a, b, acc[nt]);
    }
  }

#pragma unroll
  for (int nt = 0; nt < 8; nt++)
#pragma unroll
    for (int r = 0; r < 8; r++) {
      int trow = m0 + r + 8 * half;            // 0..8191
      int b    = trow >> 11;                   // /NT
      int t    = trow & (NT - 1);
      int e    = n_base + nt * 16 + ln;
      int h    = e >> 6;
      int d    = e & 63;
      vT[(size_t)((b * NH + h) * ND + d) * NT + t] = f2bf(acc[nt][r]);
    }
}

// ---------------- flash attention ----------------
// One wave per (b, h, 16-row q tile); 8 waves of a block share the SAME (b,h),
// so K/V S-blocks are staged cooperatively into LDS via async DMA, double-buffered.
__global__ __launch_bounds__(256) void attn_kernel(
    const u16* __restrict__ qbf,   // [NM, NE]  (e = h*64+d)
    const u16* __restrict__ kbf,   // [NM, NE]
    const u16* __restrict__ vT,    // [NB*NH*ND, NT]
    u16*       __restrict__ obf)   // [NM, NE]
{
  __shared__ __align__(16) u16 kS[2][32][64];       // 8 KB  (S rows x d)
  __shared__ __align__(16) u16 vS[2][64][32];       // 8 KB  (d rows x S)
  __shared__ __align__(16) u16 pstage[8][16][32];   // 8 KB  per-wave P transpose

  const int tid  = threadIdx.x;
  const int wave = tid >> 5;
  const int lane = tid & 31;
  const int ln   = lane & 15;
  const int half = lane >> 4;

  const int task = blockIdx.x * 8 + wave;  // 0..8191
  const int qt   = task & 127;
  const int bh   = task >> 7;
  const int b    = bh >> 4;
  const int h    = bh & 15;

  // cooperative DMA coordinates: each wave covers 4 K rows (8 chunks each)
  // and 8 V rows (4 chunks each); 2 async b128 per wave per S-block.
  const int ki = wave * 4 + (lane >> 3);   // K row within 32-block
  const int kc = (lane & 7) * 8;           // element chunk within 64-elem row
  const int dv = wave * 8 + (lane >> 2);   // V row (d) within 64
  const int vc = (lane & 3) * 8;           // element chunk within 32-elem row
  const u16* kg0 = kbf + (size_t)(b * NT + ki) * NE + h * ND + kc;
  const u16* vg0 = vT  + (size_t)((b * NH + h) * ND + dv) * NT + vc;

  // Q fragments (K = D = 64 -> two 16x32 A-frags), loaded once
  Frag aq[2];
  {
    const u16* qrow = qbf + (size_t)(b * NT + qt * 16 + ln) * NE + h * ND;
#pragma unroll
    for (int ka = 0; ka < 2; ka++) {
      const int c0 = ka * 32 + half * 8;
      aq[ka].q[0] = *(const u32x4*)(qrow + c0);
      aq[ka].q[1] = *(const u32x4*)(qrow + c0 + 16);
    }
  }

  f32x8 acc[4];
#pragma unroll
  for (int i = 0; i < 4; i++) acc[i] = zero8();
  float mrun[8], lrun[8];
#pragma unroll
  for (int r = 0; r < 8; r++) { mrun[r] = -1e30f; lrun[r] = 0.f; }

  const float scale = 0.125f;  // 1/sqrt(64)

  // prologue: stage S-block 0 into buffer 0
  async_b128((unsigned)(size_t)&kS[0][ki][kc], (unsigned long long)kg0);
  async_b128((unsigned)(size_t)&vS[0][dv][vc], (unsigned long long)vg0);

  int cur = 0;
  for (int sb = 0; sb < NSB; sb++) {
    const int s0 = sb * 32;
    if (sb + 1 < NSB) {
      // issue next block into other buffer, then retire current buffer
      async_b128((unsigned)(size_t)&kS[cur ^ 1][ki][kc],
                 (unsigned long long)(kg0 + (size_t)(sb + 1) * 32 * NE));
      async_b128((unsigned)(size_t)&vS[cur ^ 1][dv][vc],
                 (unsigned long long)(vg0 + (sb + 1) * 32));
      asm volatile("s_wait_asynccnt 2" ::: "memory");
    } else {
      asm volatile("s_wait_asynccnt 0" ::: "memory");
    }
    __syncthreads();   // staged K/V visible block-wide

    // ---- scores: two 16x16 tiles of q @ k^T from LDS ----
    f32x8 st0 = zero8(), st1 = zero8();
#pragma unroll
    for (int ka = 0; ka < 2; ka++) {
      Frag bk0, bk1;
      bk0.q[0] = *(const u32x4*)&kS[cur][ln][ka * 32 + half * 8];
      bk0.q[1] = *(const u32x4*)&kS[cur][ln][ka * 32 + 16 + half * 8];
      bk1.q[0] = *(const u32x4*)&kS[cur][16 + ln][ka * 32 + half * 8];
      bk1.q[1] = *(const u32x4*)&kS[cur][16 + ln][ka * 32 + 16 + half * 8];
      st0 = wmma_bf16(aq[ka], bk0, st0);
      st1 = wmma_bf16(aq[ka], bk1, st1);
    }
    // ---- online softmax update ----
#pragma unroll
    for (int r = 0; r < 8; r++) {
      float x0 = st0[r] * scale;
      float x1 = st1[r] * scale;
      float mb = fmaxf(x0, x1);
#pragma unroll
      for (int m = 1; m < 16; m <<= 1) mb = fmaxf(mb, __shfl_xor(mb, m, 32));
      float mn = fmaxf(mrun[r], mb);
      float al = __expf(mrun[r] - mn);
      mrun[r] = mn;
      float p0 = __expf(x0 - mn);
      float p1 = __expf(x1 - mn);
      float rsum = p0 + p1;
#pragma unroll
      for (int m = 1; m < 16; m <<= 1) rsum += __shfl_xor(rsum, m, 32);
      lrun[r] = lrun[r] * al + rsum;
      pstage[wave][r + 8 * half][ln]      = f2bf(p0);
      pstage[wave][r + 8 * half][16 + ln] = f2bf(p1);
      acc[0][r] *= al; acc[1][r] *= al; acc[2][r] *= al; acc[3][r] *= al;
    }
    asm volatile("s_wait_dscnt 0" ::: "memory");  // intra-wave LDS transpose hazard
    // ---- P as A-fragment (16x32) ----
    Frag ap;
    ap.q[0] = *(const u32x4*)&pstage[wave][ln][half * 8];
    ap.q[1] = *(const u32x4*)&pstage[wave][ln][16 + half * 8];
    // ---- acc += P @ V from LDS ----
#pragma unroll
    for (int dt = 0; dt < 4; dt++) {
      Frag bv;
      bv.q[0] = *(const u32x4*)&vS[cur][dt * 16 + ln][half * 8];
      bv.q[1] = *(const u32x4*)&vS[cur][dt * 16 + ln][16 + half * 8];
      acc[dt] = wmma_bf16(ap, bv, acc[dt]);
    }
    __syncthreads();   // everyone done with buffer `cur` before it is re-filled
    cur ^= 1;
  }

  // ---- finalize: divide by row sums, write bf16 [b,t,h,d] ----
#pragma unroll
  for (int r = 0; r < 8; r++) {
    float inv = 1.f / lrun[r];
    int t = qt * 16 + r + 8 * half;
    u16* orow = obf + (size_t)(b * NT + t) * NE + h * ND;
#pragma unroll
    for (int dt = 0; dt < 4; dt++)
      orow[dt * 16 + ln] = f2bf(acc[dt][r] * inv);
  }
}

// ---------------- output projection: attn @ Wo^T -> fp32 ----------------
__global__ __launch_bounds__(256) void proj_out_kernel(
    const u16* __restrict__ Abf,   // [NM, NE] bf16
    const u16* __restrict__ Wbf,   // [NE, NE] bf16
    float*     __restrict__ out)   // [NM, NE] fp32
{
  const int m0   = blockIdx.x * 16;
  const int tid  = threadIdx.x;
  const int wave = tid >> 5;
  const int lane = tid & 31;
  const int ln   = lane & 15;
  const int half = lane >> 4;
  const int n_base = wave * 128;

  f32x8 acc[8];
#pragma unroll
  for (int i = 0; i < 8; i++) acc[i] = zero8();

  const u16* arow = Abf + (size_t)(m0 + ln) * NE;

  for (int k0 = 0; k0 < NE; k0 += 32) {
    const int c0 = k0 + half * 8;
    __builtin_prefetch(arow + c0 + 64);
    Frag a;
    a.q[0] = *(const u32x4*)(arow + c0);
    a.q[1] = *(const u32x4*)(arow + c0 + 16);
#pragma unroll
    for (int nt = 0; nt < 8; nt++) {
      const u16* wrow = Wbf + (size_t)(n_base + nt * 16 + ln) * NE + c0;
      Frag b;
      b.q[0] = *(const u32x4*)(wrow);
      b.q[1] = *(const u32x4*)(wrow + 16);
      acc[nt] = wmma_bf16(a, b, acc[nt]);
    }
  }

#pragma unroll
  for (int nt = 0; nt < 8; nt++)
#pragma unroll
    for (int r = 0; r < 8; r++)
      out[(size_t)(m0 + r + 8 * half) * NE + n_base + nt * 16 + ln] = acc[nt][r];
}

extern "C" void kernel_launch(void* const* d_in, const int* in_sizes, int n_in,
                              void* d_out, int out_size, void* d_ws, size_t ws_size,
                              hipStream_t stream) {
  const float* xq  = (const float*)d_in[0];
  const float* xkv = (const float*)d_in[1];
  const float* Wq  = (const float*)d_in[2];
  const float* Wk  = (const float*)d_in[3];
  const float* Wv  = (const float*)d_in[4];
  const float* Wo  = (const float*)d_in[5];
  const float* qg  = (const float*)d_in[6];
  const float* qb  = (const float*)d_in[7];
  const float* kg  = (const float*)d_in[8];
  const float* kb  = (const float*)d_in[9];

  const size_t WEI = (size_t)NE * NE;   // 1M elems
  const size_t ACT = (size_t)NM * NE;   // 8M elems

  u16* Wq_bf = (u16*)d_ws;
  u16* Wk_bf = Wq_bf + WEI;
  u16* Wv_bf = Wk_bf + WEI;
  u16* Wo_bf = Wv_bf + WEI;
  u16* q_bf  = Wo_bf + WEI;
  u16* k_bf  = q_bf + ACT;
  u16* vT_bf = k_bf + ACT;
  u16* a_bf  = vT_bf + ACT;

  const int cvt_grid = (int)((WEI + 255) / 256);
  cvt_bf16_kernel<<<cvt_grid, 256, 0, stream>>>(Wq, Wq_bf, (int)WEI);
  cvt_bf16_kernel<<<cvt_grid, 256, 0, stream>>>(Wk, Wk_bf, (int)WEI);
  cvt_bf16_kernel<<<cvt_grid, 256, 0, stream>>>(Wv, Wv_bf, (int)WEI);
  cvt_bf16_kernel<<<cvt_grid, 256, 0, stream>>>(Wo, Wo_bf, (int)WEI);

  proj_ln_kernel<<<NM / 16, 256, 0, stream>>>(xq,  Wq_bf, qg, qb, q_bf);
  proj_ln_kernel<<<NM / 16, 256, 0, stream>>>(xkv, Wk_bf, kg, kb, k_bf);
  proj_v_kernel<<<NM / 16, 256, 0, stream>>>(xkv, Wv_bf, vT_bf);

  attn_kernel<<<(NB * NH * (NT / 16)) / 8, 256, 0, stream>>>(q_bf, k_bf, vT_bf, a_bf);

  proj_out_kernel<<<NM / 16, 256, 0, stream>>>(a_bf, Wo_bf, (float*)d_out);
}
